// SliceAttention_91010357002695
// MI455X (gfx1250) — matrix operand order repaired and checked by reading, not compiled, not run on hardware
//
#include <hip/hip_runtime.h>
#include <stdint.h>

typedef unsigned short u16;
typedef __attribute__((ext_vector_type(16))) __bf16 v16bf;
typedef __attribute__((ext_vector_type(8)))  float  v8f;
typedef __attribute__((ext_vector_type(4)))  unsigned int u32x4;
typedef __attribute__((ext_vector_type(4)))  int i32x4;
typedef __attribute__((ext_vector_type(8)))  int i32x8;

#define B_    2
#define N_    32768
#define HID_  512
#define NH_   8
#define G_    64
#define HD_   64
#define TOK_  64
#define NBLK_ 512          // N_/TOK_
#define FRAGH 512          // halves per 16x32 (or 32x16) bf16 fragment (1KB)

struct U8 { uint32_t w[8]; };   // one 32-byte fragment slice per lane

#if defined(__HIP_DEVICE_COMPILE__) && __has_builtin(__builtin_amdgcn_tensor_load_to_lds) && __has_builtin(__builtin_amdgcn_s_wait_tensorcnt)
#define HAVE_TDM 1
#endif

__device__ __forceinline__ u16 f2bf(float f){
  union { float f; uint32_t u; } v; v.f = f;
  uint32_t u = v.u;
  return (u16)((u + 0x7FFFu + ((u >> 16) & 1u)) >> 16);
}
// A-fragment K mapping (16-bit A 16x32): VGPR r, lane-half, low/high half e
__device__ __forceinline__ void kinvA(int k, int& e, int& hf, int& r){
  e = k & 1; hf = (k >> 4) & 1; r = (k >> 1) & 7;
}
// B-fragment K mapping (16-bit B 32x16): lanes0-15 K=0..15, lanes16-31 K=16..31
__device__ __forceinline__ void kinvB(int k, int& e, int& hf, int& r){
  e = k & 1; hf = (k >> 4) & 1; r = (k >> 1) & 7;
}
__device__ __forceinline__ int kmapB0(int r, int hf){ return 2*r + 16*hf; }

__device__ __forceinline__ v8f vzero(){
  v8f z;
  #pragma unroll
  for (int i = 0; i < 8; ++i) z[i] = 0.f;
  return z;
}
__device__ __forceinline__ v16bf ldfrag(const u16* p){
  U8 t = *(const U8*)p;
  return __builtin_bit_cast(v16bf, t);
}
__device__ __forceinline__ v8f wmma_bf16(v16bf a, v16bf b, v8f c){
  return __builtin_amdgcn_wmma_f32_16x16x32_bf16(false, a, false, b, (short)0, c, false, false);
}
__device__ __forceinline__ void gAtomicAdd(float* p, float v){
#if defined(__HIP_PLATFORM_AMD__) && defined(__HIP_DEVICE_COMPILE__)
  unsafeAtomicAdd(p, v);     // native global_atomic_add_f32
#else
  atomicAdd(p, v);
#endif
}

#ifdef HAVE_TDM
// 1-D contiguous global->LDS DMA via the Tensor Data Mover.
// nbytes must be a multiple of 8; nbytes/8 must fit in 16 bits.
__device__ __forceinline__ void tdm_load_1d(uint32_t lds_off, const void* gptr, uint32_t nbytes){
  uint64_t ga = (uint64_t)(uintptr_t)gptr;
  uint32_t elems = nbytes >> 3;                   // 8-byte elements
  u32x4 g0;
  g0[0] = 1u;                                     // count=1 valid user descriptor
  g0[1] = lds_off;                                // D#.lds_addr
  g0[2] = (uint32_t)ga;                           // D#.global_addr[31:0]
  g0[3] = (uint32_t)((ga >> 32) & 0x01FFFFFFu) | (2u << 30);  // addr[56:32] | type=2
  i32x8 g1;
  g1[0] = (int)(3u << 16);                        // workgroup_mask=0, data_size=8B
  g1[1] = (int)((elems & 0xFFFFu) << 16);         // tensor_dim0[15:0]
  g1[2] = (int)((elems >> 16) | (1u << 16));      // tensor_dim0[31:16] | tensor_dim1=1
  g1[3] = (int)((elems & 0xFFFFu) << 16);         // tile_dim0
  g1[4] = 1;                                      // tile_dim1=1
  g1[5] = (int)elems;                             // tensor_dim0_stride[31:0]
  g1[6] = 0;
  g1[7] = 0;
  i32x4 z4 = {0, 0, 0, 0};
#if __clang_major__ >= 23
  i32x8 z8 = {0, 0, 0, 0, 0, 0, 0, 0};
  __builtin_amdgcn_tensor_load_to_lds(g0, g1, z4, z4, z8, 0);
#else
  __builtin_amdgcn_tensor_load_to_lds(g0, g1, z4, z4, 0);
#endif
}
#endif

// ---------------------------------------------------------------- pack kernel
// Pre-pack W_feat / W_comb / W_out into bf16 B-fragment layout, compute b_comb.
// W_comb[o][k] = sum_c W_slice[o&63][c] * W_assign[(o>>6)*64+c][k]  (fused assign->logits)
__global__ void k_pack(const float* __restrict__ Wassign, const float* __restrict__ b_assign,
                       const float* __restrict__ Wfeat,  const float* __restrict__ Wslice,
                       const float* __restrict__ b_slice,const float* __restrict__ Wout,
                       u16* __restrict__ pWfeat, u16* __restrict__ pWcomb,
                       u16* __restrict__ pWout,  float* __restrict__ bcomb){
  const int PER = 16*32*256;  // u32 words per packed 512x512 matrix = 131072
  int tid = blockIdx.x * blockDim.x + threadIdx.x;
  if (tid < 2*PER){
    const float* W = (tid < PER) ? Wfeat : Wout;
    u16* dst       = (tid < PER) ? pWfeat : pWout;
    int u = (tid < PER) ? tid : tid - PER;
    int fragIdx = u >> 8, rem = u & 255, lane = rem >> 3, r = rem & 7;
    int kt = fragIdx >> 5, ct = fragIdx & 31;
    int k0 = kt*32 + kmapB0(r, lane >> 4);
    int n  = ct*16 + (lane & 15);
    uint32_t lo = f2bf(W[(size_t)n*HID_ + k0]);
    uint32_t hi = f2bf(W[(size_t)n*HID_ + k0 + 1]);
    ((uint32_t*)dst)[u] = lo | (hi << 16);
  } else if (tid < 3*PER){
    int u = tid - 2*PER;
    int fragIdx = u >> 8, rem = u & 255, lane = rem >> 3, r = rem & 7;
    int kt = fragIdx >> 5, ct = fragIdx & 31;
    int k0 = kt*32 + kmapB0(r, lane >> 4);
    int n  = ct*16 + (lane & 15);
    int h = n >> 6, g = n & 63;
    float s0 = 0.f, s1 = 0.f;
    for (int c = 0; c < 64; ++c){
      float wv = Wslice[g*64 + c];
      const float* wa = Wassign + (size_t)(h*64 + c)*HID_ + k0;
      s0 += wv * wa[0]; s1 += wv * wa[1];
    }
    ((uint32_t*)pWcomb)[u] = (uint32_t)f2bf(s0) | ((uint32_t)f2bf(s1) << 16);
  } else if (tid < 3*PER + 512){
    int o = tid - 3*PER;
    int h = o >> 6, g = o & 63;
    float s = b_slice[g];
    for (int c = 0; c < 64; ++c) s += Wslice[g*64 + c] * b_assign[h*64 + c];
    bcomb[o] = s;
  }
}

// ---------------------------------------------------------------- phase 1
struct P1S {
  alignas(16) u16 xA   [4*16*FRAGH];   // x block, A-frag layout        (64KB)
  alignas(16) u16 featB[2*32*FRAGH];   // feat block, B-frag (k=tok)    (64KB)
  alignas(16) u16 slwA1[8*4*2*FRAGH];  // slice_w, A-frag (m=g,k=tok)   (64KB)
  alignas(16) u16 slwA3[4*8*2*FRAGH];  // slice_w, A-frag (m=tok,k=g)   (64KB)
  float snorm[512];
  float bfeat[512];
  float bcomb[512];
  float tscale[8];
};

__global__ __launch_bounds__(256) void k_phase1(
    const float* __restrict__ x, const float* __restrict__ temperature,
    const float* __restrict__ b_feat, const u16* __restrict__ pWfeat,
    const u16* __restrict__ pWcomb, const float* __restrict__ bcomb_g,
    float* __restrict__ slice_tok, float* __restrict__ slice_norm,
    u16* __restrict__ slw_ws){
  __shared__ P1S sh;
  int b   = blockIdx.x >> 8;
  int lb  = blockIdx.x & 255;
  int tid = threadIdx.x, wave = tid >> 5, lane = tid & 31, cl = lane & 15;
  int halfsel = lane >> 4;

  for (int i = tid; i < 512; i += 256){
    sh.bfeat[i] = b_feat[i]; sh.bcomb[i] = bcomb_g[i]; sh.snorm[i] = 0.f;
  }
  if (tid < 8){
    float t = temperature[tid];
    t = fminf(fmaxf(t, 0.1f), 5.f);
    sh.tscale[tid] = 1.f / t;
  }
  v8f eacc[16];
  #pragma unroll
  for (int j = 0; j < 16; ++j) eacc[j] = vzero();
  __syncthreads();

  int mt = wave & 3;
  bool isFeat = wave < 4;
  const u16* packW = isFeat ? pWfeat : pWcomb;

  for (int blk = lb; blk < NBLK_; blk += 256){
    // ---- stage x block as bf16 A-fragments
    const float* xp = x + ((size_t)b*N_ + (size_t)blk*TOK_)*HID_;
    for (int i = tid; i < TOK_*HID_/4; i += 256){
      float4 f = ((const float4*)xp)[i];
      int tok = i >> 7, k4 = (i & 127) * 4;
      uint32_t p0 = (uint32_t)f2bf(f.x) | ((uint32_t)f2bf(f.y) << 16);
      uint32_t p1 = (uint32_t)f2bf(f.z) | ((uint32_t)f2bf(f.w) << 16);
      int tm = tok >> 4;
      int e, hf, r;
      kinvA(k4 & 31, e, hf, r);
      *(uint32_t*)&sh.xA[((tm*16 + (k4>>5))*FRAGH) + (hf*16 + (tok&15))*16 + r*2] = p0;
      kinvA((k4+2) & 31, e, hf, r);
      *(uint32_t*)&sh.xA[((tm*16 + ((k4+2)>>5))*FRAGH) + (hf*16 + (tok&15))*16 + r*2] = p1;
    }
    __syncthreads();

    // ---- prefetch next block's x stream into cache while we compute
    if (blk + 256 < NBLK_){
      const char* nx = (const char*)(x + ((size_t)b*N_ + (size_t)(blk + 256)*TOK_)*HID_);
      __builtin_prefetch(nx + (size_t)tid*512, 0, 1);
      __builtin_prefetch(nx + (size_t)tid*512 + 256, 0, 1);
    }

    // ---- dual GEMM: waves 0-3 feat, waves 4-7 fused logits (one head per group)
    for (int grp = 0; grp < 8; ++grp){
      v8f acc[4];
      #pragma unroll
      for (int j = 0; j < 4; ++j) acc[j] = vzero();
      for (int kt = 0; kt < 16; ++kt){
        v16bf a = ldfrag(&sh.xA[(mt*16 + kt)*FRAGH + lane*16]);
        #pragma unroll
        for (int j = 0; j < 4; ++j){
          int ct = grp*4 + j;
          v16bf bb = ldfrag(packW + ((size_t)(kt*32 + ct))*FRAGH + lane*16);
          acc[j] = wmma_bf16(a, bb, acc[j]);
        }
      }
      if (isFeat){
        #pragma unroll
        for (int j = 0; j < 4; ++j){
          int ct = grp*4 + j;
          float bias = sh.bfeat[ct*16 + cl];
          #pragma unroll
          for (int r = 0; r < 8; ++r){
            float val = acc[j][r] + bias;
            int tok = mt*16 + r + 8*halfsel;
            int e, hf, rr; kinvB(tok & 31, e, hf, rr);
            sh.featB[((tok>>5)*32 + ct)*FRAGH + (hf*16 + cl)*16 + rr*2 + e] = f2bf(val);
          }
        }
      } else {
        int h = grp;
        float ts = sh.tscale[h];
        float rmax[8], rinv[8];
        #pragma unroll
        for (int r = 0; r < 8; ++r){
          float m = -3.4e38f;
          #pragma unroll
          for (int j = 0; j < 4; ++j){
            acc[j][r] += sh.bcomb[h*64 + j*16 + cl];
            m = fmaxf(m, acc[j][r]);
          }
          #pragma unroll
          for (int s = 1; s < 16; s <<= 1) m = fmaxf(m, __shfl_xor(m, s, 32));
          rmax[r] = m;
        }
        #pragma unroll
        for (int r = 0; r < 8; ++r){
          float sum = 0.f;
          #pragma unroll
          for (int j = 0; j < 4; ++j){
            float ev = __expf(ts * (acc[j][r] - rmax[r]));
            acc[j][r] = ev; sum += ev;
          }
          #pragma unroll
          for (int s = 1; s < 16; s <<= 1) sum += __shfl_xor(sum, s, 32);
          rinv[r] = 1.f / sum;
        }
        #pragma unroll
        for (int j = 0; j < 4; ++j){
          float cs = 0.f;
          #pragma unroll
          for (int r = 0; r < 8; ++r){ acc[j][r] *= rinv[r]; cs += acc[j][r]; }
          cs += __shfl_xor(cs, 16, 32);
          if (lane < 16) atomicAdd(&sh.snorm[h*64 + j*16 + cl], cs);
        }
        #pragma unroll
        for (int j = 0; j < 4; ++j){
          #pragma unroll
          for (int r = 0; r < 8; ++r){
            u16 pb = f2bf(acc[j][r]);
            int tokl = mt*16 + r + 8*halfsel;
            int g = j*16 + cl;
            int e, hf, rr;
            kinvA(tokl & 31, e, hf, rr);
            sh.slwA1[((h*4 + j)*2 + (tokl>>5))*FRAGH + (hf*16 + cl)*16 + rr*2 + e] = pb;
            kinvA(g & 31, e, hf, rr);
            sh.slwA3[((mt*8 + h)*2 + (g>>5))*FRAGH + (hf*16 + (tokl & 15))*16 + rr*2 + e] = pb;
          }
        }
      }
    }
    __syncthreads();

    // ---- slice_tok einsum: 128 (h,gt,ct) tile-jobs, register-persistent accum
    #pragma unroll
    for (int jj = 0; jj < 16; ++jj){
      int flat = wave*16 + jj;
      int h = flat >> 4, gt = (flat >> 2) & 3, ct2 = flat & 3;
      #pragma unroll
      for (int kt = 0; kt < 2; ++kt){
        v16bf a  = ldfrag(&sh.slwA1[((h*4 + gt)*2 + kt)*FRAGH + lane*16]);
        v16bf bb = ldfrag(&sh.featB[(kt*32 + h*4 + ct2)*FRAGH + lane*16]);
        eacc[jj] = wmma_bf16(a, bb, eacc[jj]);
      }
    }
    // ---- spill slice_w fragments (phase-3 orientation) to global, coalesced
    {
      u16* dst = slw_ws + ((size_t)(b*NBLK_ + blk))*(size_t)(64*FRAGH);
      const U8* s8 = (const U8*)sh.slwA3;
      U8* d8 = (U8*)dst;
      for (int i = tid; i < 64*FRAGH/16; i += 256) d8[i] = s8[i];
    }
    __syncthreads();
  }

  // ---- final reduction to global (native f32 atomics)
  #pragma unroll
  for (int jj = 0; jj < 16; ++jj){
    int flat = wave*16 + jj;
    int h = flat >> 4, gt = (flat >> 2) & 3, ct2 = flat & 3;
    #pragma unroll
    for (int r = 0; r < 8; ++r){
      int g = gt*16 + r + 8*halfsel;
      int c = ct2*16 + cl;
      gAtomicAdd(&slice_tok[(((size_t)b*NH_ + h)*G_ + g)*HD_ + c], eacc[jj][r]);
    }
  }
  for (int i = tid; i < 512; i += 256) gAtomicAdd(&slice_norm[b*512 + i], sh.snorm[i]);
}

// ---------------------------------------------------------------- phase 2 (tiny)
__global__ __launch_bounds__(256) void k_phase2(
    const float* __restrict__ Wq, const float* __restrict__ Wk, const float* __restrict__ Wv,
    const float* __restrict__ slice_tok, const float* __restrict__ slice_norm,
    u16* __restrict__ outB){
  __shared__ float st[4096], qm[4096], km2[4096], vm[4096], at[4096], sn[64];
  int bh = blockIdx.x, b = bh >> 3, h = bh & 7;
  int tid = threadIdx.x;
  if (tid < 64) sn[tid] = 1.f / (slice_norm[b*512 + h*64 + tid] + 1e-6f);
  __syncthreads();
  for (int i = tid; i < 4096; i += 256) st[i] = slice_tok[(size_t)bh*4096 + i] * sn[i>>6];
  __syncthreads();
  for (int i = tid; i < 4096; i += 256){
    int g = i >> 6, c = i & 63;
    float aq = 0.f, ak = 0.f, av = 0.f;
    for (int k = 0; k < 64; ++k){
      float s = st[g*64 + k];
      aq += s*Wq[c*64 + k]; ak += s*Wk[c*64 + k]; av += s*Wv[c*64 + k];
    }
    qm[i] = aq; km2[i] = ak; vm[i] = av;
  }
  __syncthreads();
  for (int i = tid; i < 4096; i += 256){
    int g = i >> 6, j = i & 63;
    float d = 0.f;
    for (int k = 0; k < 64; ++k) d += qm[g*64 + k]*km2[j*64 + k];
    at[i] = d * 0.125f;
  }
  __syncthreads();
  if (tid < 64){
    float m = -3.4e38f;
    for (int j = 0; j < 64; ++j) m = fmaxf(m, at[tid*64 + j]);
    float s = 0.f;
    for (int j = 0; j < 64; ++j){ float e = __expf(at[tid*64 + j] - m); at[tid*64 + j] = e; s += e; }
    float inv = 1.f / s;
    for (int j = 0; j < 64; ++j) at[tid*64 + j] *= inv;
  }
  __syncthreads();
  for (int i = tid; i < 4096; i += 256){
    int g = i >> 6, c = i & 63;
    float o = 0.f;
    for (int j = 0; j < 64; ++j) o += at[g*64 + j]*vm[j*64 + c];
    int e, hf, r; kinvB(g & 31, e, hf, r);
    outB[(((size_t)bh*2 + (g>>5))*4 + (c>>4))*FRAGH + (hf*16 + (c&15))*16 + r*2 + e] = f2bf(o);
  }
}

// ---------------------------------------------------------------- phase 3
struct P3S {
  alignas(16) u16 osB [8*2*4*FRAGH];   // out_slice B-frags, all heads (64KB)
  alignas(16) u16 tokA[4*16*FRAGH];    // out_tok, A-frag layout       (64KB)
  float bout[512];
};
__global__ __launch_bounds__(256) void k_phase3(
    const u16* __restrict__ slw_ws, const u16* __restrict__ outB,
    const u16* __restrict__ pWout,  const float* __restrict__ b_out,
    float* __restrict__ out){
  __shared__ P3S sh;
  int b = blockIdx.x >> 9, blk = blockIdx.x & 511;
  int tid = threadIdx.x, wave = tid >> 5, lane = tid & 31, cl = lane & 15;
  int halfsel = lane >> 4;
#ifdef HAVE_TDM
  // Tensor Data Mover: DMA the 64KB out_slice fragment block into LDS.
  if (wave == 0){
    tdm_load_1d((uint32_t)(uintptr_t)(void*)sh.osB, outB + (size_t)b*32768, 65536);
    __builtin_amdgcn_s_wait_tensorcnt(0);
  }
  for (int i = tid; i < 512; i += 256) sh.bout[i] = b_out[i];
#else
  {
    const U8* s8 = (const U8*)(outB + (size_t)b*32768);
    U8* d8 = (U8*)sh.osB;
    for (int i = tid; i < 2048; i += 256) d8[i] = s8[i];
    for (int i = tid; i < 512; i += 256) sh.bout[i] = b_out[i];
  }
#endif
  __syncthreads();

  // part 1: out_tok = slice_w @ out_slice  (per head)
  const u16* slwblk = slw_ws + ((size_t)(b*NBLK_ + blk))*(size_t)(64*FRAGH);
  for (int jj = 0; jj < 4; ++jj){
    int flat = wave*4 + jj;
    int mt = flat & 3, h = flat >> 2;
    v16bf a0 = ldfrag(slwblk + ((size_t)((mt*8 + h)*2 + 0))*FRAGH + lane*16);
    v16bf a1 = ldfrag(slwblk + ((size_t)((mt*8 + h)*2 + 1))*FRAGH + lane*16);
    #pragma unroll
    for (int ct = 0; ct < 4; ++ct){
      v8f acc = vzero();
      v16bf b0 = ldfrag(&sh.osB[((h*2 + 0)*4 + ct)*FRAGH + lane*16]);
      v16bf b1 = ldfrag(&sh.osB[((h*2 + 1)*4 + ct)*FRAGH + lane*16]);
      acc = wmma_bf16(a0, b0, acc);
      acc = wmma_bf16(a1, b1, acc);
      #pragma unroll
      for (int r = 0; r < 8; ++r){
        int tok = mt*16 + r + 8*halfsel;
        int col = h*64 + ct*16 + cl;
        int e, hf, rr; kinvA(col & 31, e, hf, rr);
        sh.tokA[(mt*16 + (col>>5))*FRAGH + (hf*16 + (tok&15))*16 + rr*2 + e] = f2bf(acc[r]);
      }
    }
  }
  __syncthreads();

  // part 2: out = out_tok @ W_out.T + b_out
  int mt = wave >> 1, chalf = wave & 1;
  float* op = out + ((size_t)b*N_ + (size_t)blk*TOK_)*HID_;
  for (int grp = 0; grp < 4; ++grp){
    v8f acc[4];
    #pragma unroll
    for (int j = 0; j < 4; ++j) acc[j] = vzero();
    for (int kt = 0; kt < 16; ++kt){
      v16bf a = ldfrag(&sh.tokA[(mt*16 + kt)*FRAGH + lane*16]);
      #pragma unroll
      for (int j = 0; j < 4; ++j){
        int ct = chalf*16 + grp*4 + j;
        v16bf bb = ldfrag(pWout + ((size_t)(kt*32 + ct))*FRAGH + lane*16);
        acc[j] = wmma_bf16(a, bb, acc[j]);
      }
    }
    #pragma unroll
    for (int j = 0; j < 4; ++j){
      int ct = chalf*16 + grp*4 + j;
      float bias = sh.bout[ct*16 + cl];
      #pragma unroll
      for (int r = 0; r < 8; ++r){
        int tok = mt*16 + r + 8*halfsel;
        op[(size_t)tok*HID_ + ct*16 + cl] = acc[j][r] + bias;
      }
    }
  }
}

// ---------------------------------------------------------------- launcher
extern "C" void kernel_launch(void* const* d_in, const int* in_sizes, int n_in,
                              void* d_out, int out_size, void* d_ws, size_t ws_size,
                              hipStream_t stream){
  (void)in_sizes; (void)n_in; (void)out_size; (void)ws_size;
  const float* x      = (const float*)d_in[0];
  const float* temp   = (const float*)d_in[1];
  const float* Wass   = (const float*)d_in[2];
  const float* bass   = (const float*)d_in[3];
  const float* Wfeat  = (const float*)d_in[4];
  const float* bfeat  = (const float*)d_in[5];
  const float* Wslice = (const float*)d_in[6];
  const float* bslice = (const float*)d_in[7];
  const float* Wq     = (const float*)d_in[8];
  const float* Wk     = (const float*)d_in[9];
  const float* Wv     = (const float*)d_in[10];
  const float* Wout   = (const float*)d_in[11];
  const float* bout   = (const float*)d_in[12];
  float* out = (float*)d_out;

  char* ws = (char*)d_ws;
  u16*   pWfeat     = (u16*)  (ws + 0);         // 512KB
  u16*   pWcomb     = (u16*)  (ws + 524288);    // 512KB
  u16*   pWout      = (u16*)  (ws + 1048576);   // 512KB
  float* bcomb      = (float*)(ws + 1572864);   // 2KB
  float* slice_tok  = (float*)(ws + 1574912);   // 256KB
  float* slice_norm = (float*)(ws + 1837056);   // 4KB
  u16*   outB       = (u16*)  (ws + 1841152);   // 128KB
  u16*   slw_ws     = (u16*)  (ws + 1972224);   // 64MB

  hipMemsetAsync(slice_tok, 0, (size_t)B_*NH_*G_*HD_*sizeof(float), stream);
  hipMemsetAsync(slice_norm, 0, (size_t)B_*NH_*G_*sizeof(float), stream);

  k_pack<<<(3*131072 + 512 + 255)/256, 256, 0, stream>>>(
      Wass, bass, Wfeat, Wslice, bslice, Wout, pWfeat, pWcomb, pWout, bcomb);
  k_phase1<<<512, 256, 0, stream>>>(
      x, temp, bfeat, pWfeat, pWcomb, bcomb, slice_tok, slice_norm, slw_ws);
  k_phase2<<<16, 256, 0, stream>>>(Wq, Wk, Wv, slice_tok, slice_norm, outB);
  k_phase3<<<1024, 256, 0, stream>>>(slw_ws, outB, pWout, bout, out);
}
